// TwoCoAttention_32409823216336
// MI455X (gfx1250) — matrix operand order repaired and checked by reading, not compiled, not run on hardware
//
#include <hip/hip_runtime.h>

// ---------------------------------------------------------------------------
// TwoCoAttention on MI455X (gfx1250): all GEMM stages via v_wmma_f32_16x16x32_bf16
// bf16 operands pre-converted once; inner loops = b128 loads (immediate offsets)
// + wmma. 2x4 frags per wave (spill-free register budget).
// ---------------------------------------------------------------------------

typedef __attribute__((ext_vector_type(16))) __bf16 v16bf;
typedef __attribute__((ext_vector_type(8)))  __bf16 v8bf;
typedef __attribute__((ext_vector_type(4)))  __bf16 v4bf;
typedef __attribute__((ext_vector_type(8)))  float  v8f;
typedef __attribute__((ext_vector_type(4)))  float  v4f;

#define CIN   2048
#define CHID  256
#define PSP   1024   // 32*32 spatial positions
#define NB    8
#define NPROB 16     // 2 directions * 8 batches

#define WMMA_BF16(A, B, C) \
  __builtin_amdgcn_wmma_f32_16x16x32_bf16(false, (A), false, (B), (short)0, (C), false, false)

static __device__ __forceinline__ v8f zero8() {
  v8f z;
  #pragma unroll
  for (int i = 0; i < 8; ++i) z[i] = 0.0f;
  return z;
}

// A-fragment (16xK slice, 16-bit): lane half h holds K = {k0+8h..+7} U {k0+16+8h..+7}
static __device__ __forceinline__ v16bf frag_a_bf16(const __bf16* rowp, int h) {
  v8bf r0 = *(const v8bf*)(rowp + h * 8);
  v8bf r1 = *(const v8bf*)(rowp + 16 + h * 8);
  v16bf o;
  #pragma unroll
  for (int e = 0; e < 8; ++e) { o[e] = r0[e]; o[e + 8] = r1[e]; }
  return o;
}

// B-fragment (Kx16 slice, 16-bit): lane half h holds K = {k0+8h..+7} U {k0+16+8h..+7}
static __device__ __forceinline__ v16bf frag_b_bf16(const __bf16* rowp, int h) {
  v8bf r0 = *(const v8bf*)(rowp + h * 8);
  v8bf r1 = *(const v8bf*)(rowp + 16 + h * 8);
  v16bf o;
  #pragma unroll
  for (int e = 0; e < 8; ++e) { o[e] = r0[e]; o[e + 8] = r1[e]; }
  return o;
}

// ---------------------------------------------------------------------------
// Kernel 0: X[c][i] (f32) -> Xt[i][c] (bf16, transposed)  AND  Xc[c][i] (bf16)
// ---------------------------------------------------------------------------
__global__ void k0_cvt_layouts(const float* __restrict__ f1,
                               const float* __restrict__ f2,
                               __bf16* __restrict__ Xt,
                               __bf16* __restrict__ Xc) {
  __shared__ float tile[32][33];
  const int fb = blockIdx.z;                       // 0..7 -> f1, 8..15 -> f2
  const float* X = (fb < NB ? f1 : f2) + (size_t)(fb & 7) * CIN * PSP;
  __bf16* Yt = Xt + (size_t)fb * PSP * CIN;
  __bf16* Yc = Xc + (size_t)fb * CIN * PSP;
  const int c0 = blockIdx.x * 32, i0 = blockIdx.y * 32;
  const int tx = threadIdx.x, ty = threadIdx.y;    // (32, 8)
  #pragma unroll
  for (int k = 0; k < 4; ++k) {
    const float v = X[(size_t)(c0 + ty + 8 * k) * PSP + i0 + tx];
    tile[ty + 8 * k][tx] = v;
    Yc[(size_t)(c0 + ty + 8 * k) * PSP + i0 + tx] = (__bf16)v;   // coalesced copy-cvt
  }
  __syncthreads();
  #pragma unroll
  for (int k = 0; k < 4; ++k)
    Yt[(size_t)(i0 + ty + 8 * k) * CIN + c0 + tx] = (__bf16)tile[tx][ty + 8 * k];
}

// ---------------------------------------------------------------------------
// Kernel W: convert the 4 weight matrices (256x2048 f32) to bf16 slabs
// slab order matches k1's mm selector: 0=Wq1, 1=Wk1, 2=Wq2, 3=Wk2
// ---------------------------------------------------------------------------
__global__ void kw_cvt_weights(const float* __restrict__ Wq1,
                               const float* __restrict__ Wk1,
                               const float* __restrict__ Wq2,
                               const float* __restrict__ Wk2,
                               __bf16* __restrict__ Wb) {
  const float* src;
  switch (blockIdx.y) {
    case 0:  src = Wq1; break;
    case 1:  src = Wk1; break;
    case 2:  src = Wq2; break;
    default: src = Wk2; break;
  }
  const size_t idx = ((size_t)blockIdx.x * blockDim.x + threadIdx.x) * 4;
  const v4f t = *(const v4f*)(src + idx);
  v4bf o;
  #pragma unroll
  for (int e = 0; e < 4; ++e) o[e] = (__bf16)t[e];
  *(v4bf*)(Wb + (size_t)blockIdx.y * CHID * CIN + idx) = o;
}

// ---------------------------------------------------------------------------
// Kernel 1: projections  Qt/Kt[i][o] = sum_c Xt[i][c] * Wb[o][c] + bias[o]
// per-wave 32x64 tile (2x4 WMMA frags), K=2048; fragment rows via imm offsets.
// ---------------------------------------------------------------------------
__global__ void k1_proj(const __bf16* __restrict__ Xt,
                        const __bf16* __restrict__ Wb,
                        const float* __restrict__ bq1, const float* __restrict__ bk1,
                        const float* __restrict__ bq2, const float* __restrict__ bk2,
                        __bf16* __restrict__ Qt, __bf16* __restrict__ Kt) {
  const size_t DIRSTRIDE = (size_t)NB * PSP * CHID;
  const int z  = blockIdx.z;         // 0..31
  const int mm = z >> 3, b = z & 7;

  const float* bias; __bf16* Out; int xsel;
  switch (mm) {
    case 0:  bias = bq1; Out = Qt;             xsel = 0; break; // dir0 Q <- F1
    case 1:  bias = bk1; Out = Kt;             xsel = 1; break; // dir0 K <- F2
    case 2:  bias = bq2; Out = Qt + DIRSTRIDE; xsel = 1; break; // dir1 Q <- F2
    default: bias = bk2; Out = Kt + DIRSTRIDE; xsel = 0; break; // dir1 K <- F1
  }
  const __bf16* W = Wb + (size_t)mm * CHID * CIN;
  const __bf16* X = Xt + (size_t)(xsel * NB + b) * PSP * CIN;
  Out += (size_t)b * PSP * CHID;

  const int lane = threadIdx.x & 31;
  const int h = lane >> 4, lr = lane & 15;
  const int w = threadIdx.x >> 5;
  const int m0 = blockIdx.x * 64 + (w >> 1) * 32;   // position rows
  const int n0 = blockIdx.y * 128 + (w & 1) * 64;   // hidden cols

  v8f acc[2][4];
  #pragma unroll
  for (int mf = 0; mf < 2; ++mf)
    #pragma unroll
    for (int nf = 0; nf < 4; ++nf) acc[mf][nf] = zero8();

  const __bf16* Abase = X + (size_t)(m0 + lr) * CIN;   // + mf*16*CIN (imm)
  const __bf16* Bbase = W + (size_t)(n0 + lr) * CIN;   // + nf*16*CIN (imm)

  for (int k0 = 0; k0 < CIN; k0 += 32) {
    v16bf a[2], bb[4];
    #pragma unroll
    for (int mf = 0; mf < 2; ++mf) a[mf]  = frag_a_bf16(Abase + mf * 16 * CIN + k0, h);
    #pragma unroll
    for (int nf = 0; nf < 4; ++nf) bb[nf] = frag_b_bf16(Bbase + nf * 16 * CIN + k0, h);
    #pragma unroll
    for (int mf = 0; mf < 2; ++mf)
      #pragma unroll
      for (int nf = 0; nf < 4; ++nf)
        acc[mf][nf] = WMMA_BF16(a[mf], bb[nf], acc[mf][nf]);
  }

  __bf16* Obase = Out + (size_t)(m0 + 8 * h) * CHID + n0 + lr;
  #pragma unroll
  for (int nf = 0; nf < 4; ++nf) {
    const float bv = bias[n0 + nf * 16 + lr];
    #pragma unroll
    for (int mf = 0; mf < 2; ++mf)
      #pragma unroll
      for (int r = 0; r < 8; ++r)
        Obase[(mf * 16 + r) * CHID + nf * 16] = (__bf16)(acc[mf][nf][r] + bv);
  }
}

// ---------------------------------------------------------------------------
// Kernel 2: S = Qt * Kt^T (K=256) + fused row softmax -> P (bf16)
// One workgroup per (problem, 16-row i-block); 4 waves x 256-wide j-strips.
// ---------------------------------------------------------------------------
__global__ void k2_attn(const __bf16* __restrict__ Qt,
                        const __bf16* __restrict__ Kt,
                        __bf16* __restrict__ Pm) {
  __shared__ float red[4][16];
  const int z  = blockIdx.y;
  const int i0 = blockIdx.x * 16;
  const __bf16* Q = Qt + (size_t)z * PSP * CHID;
  const __bf16* K = Kt + (size_t)z * PSP * CHID;
  __bf16* P = Pm + (size_t)z * PSP * PSP;

  const int lane = threadIdx.x & 31;
  const int h = lane >> 4, lr = lane & 15;
  const int w = threadIdx.x >> 5;
  const int jw = w * 256;

  v8f acc[16];
  #pragma unroll
  for (int jf = 0; jf < 16; ++jf) acc[jf] = zero8();

  const __bf16* Abase = Q + (size_t)(i0 + lr) * CHID;
  const __bf16* Bbase = K + (size_t)(jw + lr) * CHID;  // + jf*16*CHID (imm)

  for (int k0 = 0; k0 < CHID; k0 += 32) {
    const v16bf a = frag_a_bf16(Abase + k0, h);
    #pragma unroll
    for (int jf = 0; jf < 16; ++jf) {
      const v16bf bb = frag_b_bf16(Bbase + jf * 16 * CHID + k0, h);
      acc[jf] = WMMA_BF16(a, bb, acc[jf]);
    }
  }

  // --- row max (rows i0 + r + 8h, r = 0..7 per lane) ---
  float mx[8];
  #pragma unroll
  for (int r = 0; r < 8; ++r) {
    float m = acc[0][r];
    #pragma unroll
    for (int jf = 1; jf < 16; ++jf) m = fmaxf(m, acc[jf][r]);
    mx[r] = m;
  }
  #pragma unroll
  for (int d = 1; d <= 8; d <<= 1)
    #pragma unroll
    for (int r = 0; r < 8; ++r) mx[r] = fmaxf(mx[r], __shfl_xor(mx[r], d, 32));
  if (lr == 0) {
    #pragma unroll
    for (int r = 0; r < 8; ++r) red[w][8 * h + r] = mx[r];
  }
  __syncthreads();
  #pragma unroll
  for (int r = 0; r < 8; ++r) {
    float m = red[0][8 * h + r];
    #pragma unroll
    for (int w2 = 1; w2 < 4; ++w2) m = fmaxf(m, red[w2][8 * h + r]);
    mx[r] = m;
  }
  __syncthreads();

  // --- exp + row sum ---
  float sm[8];
  #pragma unroll
  for (int r = 0; r < 8; ++r) sm[r] = 0.0f;
  #pragma unroll
  for (int jf = 0; jf < 16; ++jf)
    #pragma unroll
    for (int r = 0; r < 8; ++r) {
      const float e = __expf(acc[jf][r] - mx[r]);
      acc[jf][r] = e;
      sm[r] += e;
    }
  #pragma unroll
  for (int d = 1; d <= 8; d <<= 1)
    #pragma unroll
    for (int r = 0; r < 8; ++r) sm[r] += __shfl_xor(sm[r], d, 32);
  if (lr == 0) {
    #pragma unroll
    for (int r = 0; r < 8; ++r) red[w][8 * h + r] = sm[r];
  }
  __syncthreads();
  float inv[8];
  #pragma unroll
  for (int r = 0; r < 8; ++r) {
    const float s = red[0][8 * h + r] + red[1][8 * h + r] +
                    red[2][8 * h + r] + red[3][8 * h + r];
    inv[r] = 1.0f / s;
  }

  // --- normalized probabilities, bf16 ---
  __bf16* Pbase = P + (size_t)(i0 + 8 * h) * PSP + jw + lr;
  #pragma unroll
  for (int jf = 0; jf < 16; ++jf)
    #pragma unroll
    for (int r = 0; r < 8; ++r)
      Pbase[r * PSP + jf * 16] = (__bf16)(acc[jf][r] * inv[r]);
}

// ---------------------------------------------------------------------------
// Kernel 3: Out[c][i] = sum_j Xc[c][j] * P[i][j]   (K=1024)
// per-wave 32x64 tile (2x4 frags); all-bf16 operands, f32 output.
// ---------------------------------------------------------------------------
__global__ void k3_attended(const __bf16* __restrict__ Xc,
                            const __bf16* __restrict__ Pm,
                            float* __restrict__ out) {
  const int z = blockIdx.z, dir = z >> 3, b = z & 7;
  // reference features: dir0 -> f2 (slab 8+b), dir1 -> f1 (slab b)
  const __bf16* V = Xc + (size_t)((dir == 0 ? NB : 0) + b) * CIN * PSP;
  const __bf16* P = Pm + (size_t)z * PSP * PSP;
  float* O = out + (size_t)z * CIN * PSP;   // z = dir*8+b matches concatenated (x1, x2)

  const int lane = threadIdx.x & 31;
  const int h = lane >> 4, lr = lane & 15;
  const int w = threadIdx.x >> 5;
  const int m0 = blockIdx.x * 64 + (w >> 1) * 32;   // channel rows
  const int n0 = blockIdx.y * 128 + (w & 1) * 64;   // position cols

  v8f acc[2][4];
  #pragma unroll
  for (int mf = 0; mf < 2; ++mf)
    #pragma unroll
    for (int nf = 0; nf < 4; ++nf) acc[mf][nf] = zero8();

  const __bf16* Abase = V + (size_t)(m0 + lr) * PSP;   // + mf*16*PSP (imm)
  const __bf16* Bbase = P + (size_t)(n0 + lr) * PSP;   // + nf*16*PSP (imm)

  for (int k0 = 0; k0 < PSP; k0 += 32) {
    v16bf a[2], bb[4];
    #pragma unroll
    for (int mf = 0; mf < 2; ++mf) a[mf]  = frag_a_bf16(Abase + mf * 16 * PSP + k0, h);
    #pragma unroll
    for (int nf = 0; nf < 4; ++nf) bb[nf] = frag_b_bf16(Bbase + nf * 16 * PSP + k0, h);
    #pragma unroll
    for (int mf = 0; mf < 2; ++mf)
      #pragma unroll
      for (int nf = 0; nf < 4; ++nf)
        acc[mf][nf] = WMMA_BF16(a[mf], bb[nf], acc[mf][nf]);
  }

  float* Obase = O + (size_t)(m0 + 8 * h) * PSP + n0 + lr;
  #pragma unroll
  for (int mf = 0; mf < 2; ++mf)
    #pragma unroll
    for (int nf = 0; nf < 4; ++nf)
      #pragma unroll
      for (int r = 0; r < 8; ++r)
        Obase[(mf * 16 + r) * PSP + nf * 16] = acc[mf][nf][r];
}

// ---------------------------------------------------------------------------
extern "C" void kernel_launch(void* const* d_in, const int* in_sizes, int n_in,
                              void* d_out, int out_size, void* d_ws, size_t ws_size,
                              hipStream_t stream) {
  (void)in_sizes; (void)n_in; (void)out_size; (void)ws_size;
  const float* f1  = (const float*)d_in[0];
  const float* f2  = (const float*)d_in[1];
  const float* Wq1 = (const float*)d_in[2]; const float* bq1 = (const float*)d_in[3];
  const float* Wk1 = (const float*)d_in[4]; const float* bk1 = (const float*)d_in[5];
  const float* Wq2 = (const float*)d_in[6]; const float* bq2 = (const float*)d_in[7];
  const float* Wk2 = (const float*)d_in[8]; const float* bk2 = (const float*)d_in[9];
  float* out = (float*)d_out;

  // Workspace layout (bf16 elements):
  //   Xt [16][1024][2048]  64 MB | Xc [16][2048][1024]  64 MB
  //   Qt [2][8][1024][256]  8 MB | Kt 8 MB | Pm [16][1024][1024] 32 MB | Wb 4 MB
  char* ws = (char*)d_ws;
  const size_t XSLAB  = (size_t)NPROB * PSP * CIN;
  const size_t QKSLAB = (size_t)NPROB * PSP * CHID;
  __bf16* Xt = (__bf16*)ws;
  __bf16* Xc = Xt + XSLAB;
  __bf16* Qt = Xc + XSLAB;
  __bf16* Kt = Qt + QKSLAB;
  __bf16* Pm = Kt + QKSLAB;
  __bf16* Wb = Pm + (size_t)NPROB * PSP * PSP;

  k0_cvt_layouts<<<dim3(CIN / 32, PSP / 32, NPROB), dim3(32, 8), 0, stream>>>(f1, f2, Xt, Xc);
  kw_cvt_weights<<<dim3((CHID * CIN) / (256 * 4), 4), dim3(256), 0, stream>>>(
      Wq1, Wk1, Wq2, Wk2, Wb);
  k1_proj<<<dim3(PSP / 64, CHID / 128, 32), dim3(128), 0, stream>>>(
      Xt, Wb, bq1, bk1, bq2, bk2, Qt, Kt);
  k2_attn<<<dim3(PSP / 16, NPROB), dim3(128), 0, stream>>>(Qt, Kt, Pm);
  k3_attended<<<dim3(CIN / 64, PSP / 128, NPROB), dim3(128), 0, stream>>>(Xc, Pm, out);
}
